// ModelQK_13357348290921
// MI455X (gfx1250) — matrix operand order repaired and checked by reading, not compiled, not run on hardware
//
#include <hip/hip_runtime.h>
#include <hip/hip_bf16.h>

typedef _Float16 h16 __attribute__((ext_vector_type(16)));
typedef _Float16 h8  __attribute__((ext_vector_type(8)));
typedef float    f8  __attribute__((ext_vector_type(8)));

#define QDIM  192
#define KDIM  64
#define CDIM  256
#define NNODE 10000
#define NEDGE 320000
#define NMASK 5000
#define DOUT  128
#define YC    640      // [q(256) | k(256) | v64(64) | s64(64)]
#define MPAD  5008     // 313*16

__device__ __forceinline__ unsigned enc_f(float f) {
  unsigned s = __float_as_uint(f);
  return (s & 0x80000000u) ? ~s : (s | 0x80000000u);
}
__device__ __forceinline__ float dec_f(unsigned e) {
  unsigned s = (e & 0x80000000u) ? (e ^ 0x80000000u) : ~e;
  return __uint_as_float(s);
}

// ---- prep: concatenated transposed f16 weights + bias ----
__global__ void k_build_wcat(const float* __restrict__ Wq, const float* __restrict__ Wk,
                             const float* __restrict__ Wv, const float* __restrict__ Ws,
                             const float* __restrict__ bq, const float* __restrict__ bk,
                             const float* __restrict__ bv, const float* __restrict__ bs,
                             _Float16* __restrict__ wt, float* __restrict__ bcat) {
  int idx = blockIdx.x * blockDim.x + threadIdx.x;
  if (idx < YC * CDIM) {
    int n = idx / CDIM, k = idx % CDIM;
    float w;
    if      (n < 256) w = Wq[k * CDIM + n];
    else if (n < 512) w = Wk[k * CDIM + (n - 256)];
    else if (n < 576) w = Wv[k * CDIM + (192 + n - 512)];
    else              w = Ws[k * CDIM + (192 + n - 576)];
    wt[n * CDIM + k] = (_Float16)w;
  }
  if (idx < YC) {
    int n = idx;
    float b = (n < 256) ? bq[n] : (n < 512) ? bk[n - 256]
            : (n < 576) ? bv[192 + n - 512] : bs[192 + n - 576];
    bcat[n] = b;
  }
}

__global__ void k_build_wo(const float* __restrict__ Wo, _Float16* __restrict__ wot) {
  int idx = blockIdx.x * blockDim.x + threadIdx.x;
  if (idx >= DOUT * KDIM) return;
  int n = idx / KDIM, k = idx % KDIM;
  wot[n * KDIM + k] = (_Float16)Wo[k * DOUT + n];
}

// ---- per-call state init (deterministic every launch) ----
__global__ void k_init(unsigned* __restrict__ menc, float* __restrict__ denom,
                       int* __restrict__ in_mask, float* __restrict__ agg) {
  int i = blockIdx.x * blockDim.x + threadIdx.x;
  if (i >= NNODE * 64) return;
  agg[i] = 0.0f;
  if ((i & 63) == 0) {
    int n = i >> 6;
    menc[n]    = 0x007FFFFFu;  // enc(-inf)
    denom[n]   = 0.0f;
    in_mask[n] = 0;
  }
}

__global__ void k_set_mask(const int* __restrict__ mask_idx, int* __restrict__ in_mask) {
  int i = blockIdx.x * blockDim.x + threadIdx.x;
  if (i < NMASK) in_mask[mask_idx[i]] = 1;
}

// ---- build x = [query | one_hot(key_idx)] as f16 ----
__global__ void k_build_x(const float* __restrict__ query, const int* __restrict__ key_idx,
                          _Float16* __restrict__ x) {
  int idx = blockIdx.x * blockDim.x + threadIdx.x;
  if (idx >= NNODE * CDIM) return;
  int r = idx / CDIM, c = idx % CDIM;
  float v = (c < QDIM) ? query[r * QDIM + c]
                       : ((c - QDIM) == key_idx[r] ? 1.0f : 0.0f);
  x[idx] = (_Float16)v;
}

// ---- fused node GEMM: Y(10000x640) = x(10000x256) @ Wcat + bcat, via WMMA f16->f32 ----
__global__ void k_gemm_nodes(const _Float16* __restrict__ x, const _Float16* __restrict__ wt,
                             const float* __restrict__ bcat, float* __restrict__ Y) {
  int wid  = (blockIdx.x * blockDim.x + threadIdx.x) >> 5;
  int lane = threadIdx.x & 31;
  if (wid >= 625 * 10) return;              // uniform per wave
  int mtile = wid / 10, nstrip = wid % 10;
  int m = lane & 15, hi = lane >> 4;
  const _Float16* arow = x + (size_t)(mtile * 16 + m) * CDIM;

  f8 acc[4] = {};
  for (int k8 = 0; k8 < 8; ++k8) {
    int kb = k8 * 32;
    h8 alo = *(const h8*)(arow + kb + hi * 16);
    h8 ahi = *(const h8*)(arow + kb + hi * 16 + 8);
    h16 a;
#pragma unroll
    for (int i = 0; i < 8; ++i) { a[i] = alo[i]; a[8 + i] = ahi[i]; }
#pragma unroll
    for (int j = 0; j < 4; ++j) {
      int n = nstrip * 64 + j * 16 + m;
      h16 b = *(const h16*)(wt + (size_t)n * CDIM + kb + hi * 16);
      acc[j] = __builtin_amdgcn_wmma_f32_16x16x32_f16(false, a, false, b,
                                                      (short)0, acc[j], false, false);
    }
  }
#pragma unroll
  for (int j = 0; j < 4; ++j) {
    int col = nstrip * 64 + j * 16 + m;
    float bia = bcat[col];
#pragma unroll
    for (int r = 0; r < 8; ++r) {
      int row = mtile * 16 + hi * 8 + r;
      Y[(size_t)row * YC + col] = acc[j][r] + bia;
    }
  }
}

// ---- edge pass 1: score = dot(q[dst], k[src])/16, atomic segment max ----
__global__ void k_scores(const int* __restrict__ ei, const int* __restrict__ in_mask,
                         const float* __restrict__ Y, float* __restrict__ score,
                         unsigned* __restrict__ menc) {
  int wid  = (blockIdx.x * blockDim.x + threadIdx.x) >> 5;
  int lane = threadIdx.x & 31;
  if (wid >= NEDGE) return;
  int src = ei[wid], dst = ei[NEDGE + wid];
  if (!in_mask[dst]) return;                // uniform per wave
  const float4* q = (const float4*)(Y + (size_t)dst * YC);
  const float4* k = (const float4*)(Y + (size_t)src * YC + 256);
  float4 q0 = q[lane * 2], q1 = q[lane * 2 + 1];
  float4 k0 = k[lane * 2], k1 = k[lane * 2 + 1];
  float p = q0.x * k0.x + q0.y * k0.y + q0.z * k0.z + q0.w * k0.w
          + q1.x * k1.x + q1.y * k1.y + q1.z * k1.z + q1.w * k1.w;
  for (int off = 16; off; off >>= 1) p += __shfl_down(p, off);
  if (lane == 0) {
    float s = p * (1.0f / 16.0f);           // 1/sqrt(256)
    score[wid] = s;
    atomicMax(&menc[dst], enc_f(s));
  }
}

// ---- edge pass 2: e = exp(score - m[dst]); denom[dst] += e ----
__global__ void k_expsum(const int* __restrict__ ei, const int* __restrict__ in_mask,
                         const unsigned* __restrict__ menc, float* __restrict__ score,
                         float* __restrict__ denom) {
  int e = blockIdx.x * blockDim.x + threadIdx.x;
  if (e >= NEDGE) return;
  int dst = ei[NEDGE + e];
  if (!in_mask[dst]) return;
  float m  = dec_f(menc[dst]);
  float ev = expf(score[e] - m);
  score[e] = ev;                            // reuse buffer for exp values
  atomicAdd(&denom[dst], ev);
}

// ---- edge pass 3: agg64[dst] += alpha * v64[src] (only the 64 cols that matter) ----
__global__ void k_aggregate(const int* __restrict__ ei, const int* __restrict__ in_mask,
                            const float* __restrict__ escore, const float* __restrict__ denom,
                            const float* __restrict__ Y, float* __restrict__ agg) {
  int wid  = (blockIdx.x * blockDim.x + threadIdx.x) >> 5;
  int lane = threadIdx.x & 31;
  if (wid >= NEDGE) return;
  int src = ei[wid], dst = ei[NEDGE + wid];
  if (!in_mask[dst]) return;
  float a = escore[wid] / (denom[dst] + 1e-16f);
  float2 v = *(const float2*)(Y + (size_t)src * YC + 512 + lane * 2);
  atomicAdd(&agg[(size_t)dst * 64 + lane * 2],     a * v.x);
  atomicAdd(&agg[(size_t)dst * 64 + lane * 2 + 1], a * v.y);
}

// ---- gather masked rows, ReLU, pack to zero-padded f16 (5008 x 64) ----
__global__ void k_h64(const int* __restrict__ mask_idx, const float* __restrict__ agg,
                      const float* __restrict__ Y, _Float16* __restrict__ h64) {
  int idx = blockIdx.x * blockDim.x + threadIdx.x;
  if (idx >= MPAD * 64) return;
  int i = idx >> 6, c = idx & 63;
  float v = 0.0f;
  if (i < NMASK) {
    int r = mask_idx[i];
    v = agg[(size_t)r * 64 + c] + Y[(size_t)r * YC + 576 + c];
    v = v > 0.0f ? v : 0.0f;
  }
  h64[idx] = (_Float16)v;
}

// ---- final GEMM: out(5000x128) = h64 @ Wo + bo, via WMMA ----
__global__ void k_gemm_out(const _Float16* __restrict__ h64, const _Float16* __restrict__ wot,
                           const float* __restrict__ bo, float* __restrict__ out) {
  int wid  = (blockIdx.x * blockDim.x + threadIdx.x) >> 5;
  int lane = threadIdx.x & 31;
  if (wid >= 313 * 2) return;
  int mtile = wid >> 1, nstrip = wid & 1;
  int m = lane & 15, hi = lane >> 4;
  const _Float16* arow = h64 + (size_t)(mtile * 16 + m) * KDIM;

  f8 acc[4] = {};
#pragma unroll
  for (int k8 = 0; k8 < 2; ++k8) {
    int kb = k8 * 32;
    h8 alo = *(const h8*)(arow + kb + hi * 16);
    h8 ahi = *(const h8*)(arow + kb + hi * 16 + 8);
    h16 a;
#pragma unroll
    for (int i = 0; i < 8; ++i) { a[i] = alo[i]; a[8 + i] = ahi[i]; }
#pragma unroll
    for (int j = 0; j < 4; ++j) {
      int n = nstrip * 64 + j * 16 + m;
      h16 b = *(const h16*)(wot + (size_t)n * KDIM + kb + hi * 16);
      acc[j] = __builtin_amdgcn_wmma_f32_16x16x32_f16(false, a, false, b,
                                                      (short)0, acc[j], false, false);
    }
  }
#pragma unroll
  for (int j = 0; j < 4; ++j) {
    int col = nstrip * 64 + j * 16 + m;
    float bia = bo[col];
#pragma unroll
    for (int r = 0; r < 8; ++r) {
      int row = mtile * 16 + hi * 8 + r;
      if (row < NMASK) out[(size_t)row * DOUT + col] = acc[j][r] + bia;
    }
  }
}

extern "C" void kernel_launch(void* const* d_in, const int* in_sizes, int n_in,
                              void* d_out, int out_size, void* d_ws, size_t ws_size,
                              hipStream_t stream) {
  (void)in_sizes; (void)n_in; (void)out_size; (void)ws_size;
  const float* query   = (const float*)d_in[0];
  const int*   key_idx = (const int*)d_in[1];
  const int*   ei      = (const int*)d_in[2];
  const int*   maskIdx = (const int*)d_in[3];
  const float* Wq = (const float*)d_in[4];  const float* bq = (const float*)d_in[5];
  const float* Wk = (const float*)d_in[6];  const float* bk = (const float*)d_in[7];
  const float* Wv = (const float*)d_in[8];  const float* bv = (const float*)d_in[9];
  const float* Ws = (const float*)d_in[10]; const float* bs = (const float*)d_in[11];
  const float* Wo = (const float*)d_in[12]; const float* bo = (const float*)d_in[13];
  float* out = (float*)d_out;

  char* p = (char*)d_ws;
  auto alloc = [&](size_t bytes) -> char* {
    char* r = p; p += (bytes + 255) & ~size_t(255); return r;
  };
  _Float16* x     = (_Float16*)alloc((size_t)NNODE * CDIM * 2);
  _Float16* wcat  = (_Float16*)alloc((size_t)YC * CDIM * 2);
  float*    bcat  = (float*)   alloc((size_t)YC * 4);
  float*    Y     = (float*)   alloc((size_t)NNODE * YC * 4);
  float*    score = (float*)   alloc((size_t)NEDGE * 4);
  unsigned* menc  = (unsigned*)alloc((size_t)NNODE * 4);
  float*    denom = (float*)   alloc((size_t)NNODE * 4);
  float*    agg   = (float*)   alloc((size_t)NNODE * 64 * 4);
  int*      inm   = (int*)     alloc((size_t)NNODE * 4);
  _Float16* h64   = (_Float16*)alloc((size_t)MPAD * 64 * 2);
  _Float16* wot   = (_Float16*)alloc((size_t)DOUT * KDIM * 2);

  k_build_wcat<<<(YC * CDIM + 255) / 256, 256, 0, stream>>>(Wq, Wk, Wv, Ws, bq, bk, bv, bs, wcat, bcat);
  k_build_wo  <<<(DOUT * KDIM + 255) / 256, 256, 0, stream>>>(Wo, wot);
  k_init      <<<(NNODE * 64 + 255) / 256, 256, 0, stream>>>(menc, denom, inm, agg);
  k_set_mask  <<<(NMASK + 255) / 256, 256, 0, stream>>>(maskIdx, inm);
  k_build_x   <<<(NNODE * CDIM + 255) / 256, 256, 0, stream>>>(query, key_idx, x);

  {
    int waves = 625 * 10;
    k_gemm_nodes<<<(waves * 32 + 255) / 256, 256, 0, stream>>>(x, wcat, bcat, Y);
  }
  k_scores   <<<(NEDGE * 32 + 255) / 256, 256, 0, stream>>>(ei, inm, Y, score, menc);
  k_expsum   <<<(NEDGE + 255) / 256, 256, 0, stream>>>(ei, inm, menc, score, denom);
  k_aggregate<<<(NEDGE * 32 + 255) / 256, 256, 0, stream>>>(ei, inm, score, denom, Y, agg);
  k_h64      <<<(MPAD * 64 + 255) / 256, 256, 0, stream>>>(maskIdx, agg, Y, h64);
  {
    int waves = 313 * 2;
    k_gemm_out<<<(waves * 32 + 255) / 256, 256, 0, stream>>>(h64, wot, bo, out);
  }
}
